// GIN_ESM_DTA_20907900797458
// MI455X (gfx1250) — matrix operand order repaired and checked by reading, not compiled, not run on hardware
//
#include <hip/hip_runtime.h>

typedef __attribute__((ext_vector_type(16))) _Float16 v16h;
typedef __attribute__((ext_vector_type(8)))  _Float16 v8h;
typedef __attribute__((ext_vector_type(8)))  float    v8f;
typedef __attribute__((ext_vector_type(4)))  int      v4i;

#define BN_FOLD 0.99999500003749980f /* 1/sqrt(1+1e-5) */

#if defined(__gfx1250__) && __has_builtin(__builtin_amdgcn_global_load_async_to_lds_b128)
#define HAVE_ASYNC_LDS 1
#else
#define HAVE_ASYNC_LDS 0
#endif

// ---------------------------------------------------------------- utilities
__global__ void k_zero(float* p, long long n) {
  long long t = (long long)blockIdx.x * blockDim.x + threadIdx.x;
  if (t < n) p[t] = 0.f;
}

// Transpose + convert weight: W[K x Ncols] f32 -> Wt[Ncols x Kpad] f16 (zero pad K..Kpad)
__global__ void k_wt(const float* __restrict__ W, _Float16* __restrict__ Wt,
                     int K, int Ncols, int Kpad) {
  long long t = (long long)blockIdx.x * blockDim.x + threadIdx.x;
  if (t >= (long long)Ncols * Kpad) return;
  int n = (int)(t / Kpad), k = (int)(t % Kpad);
  Wt[t] = (k < K) ? (_Float16)W[(long long)k * Ncols + n] : (_Float16)0.f;
}

// scalar scatter-add (F not multiple of 4): agg[dst] += h[src]
__global__ void k_scatter1(const float* __restrict__ h, const int* __restrict__ src,
                           const int* __restrict__ dst, float* agg, long long E, int F) {
  long long t = (long long)blockIdx.x * blockDim.x + threadIdx.x;
  if (t >= E * F) return;
  long long e = t / F; int f = (int)(t % F);
  atomicAdd(&agg[(long long)dst[e] * F + f], h[(long long)src[e] * F + f]);
}

// vec4 scatter-add (F multiple of 4)
__global__ void k_scatter4(const float* __restrict__ h, const int* __restrict__ src,
                           const int* __restrict__ dst, float* agg, long long E, int F) {
  long long t = (long long)blockIdx.x * blockDim.x + threadIdx.x;
  int ch = F >> 2;
  if (t >= E * ch) return;
  long long e = t / ch; int c = (int)(t % ch) * 4;
  const float* hp = h + (long long)src[e] * F + c;
  float*       ap = agg + (long long)dst[e] * F + c;
  atomicAdd(ap + 0, hp[0]); atomicAdd(ap + 1, hp[1]);
  atomicAdd(ap + 2, hp[2]); atomicAdd(ap + 3, hp[3]);
}

// global_add_pool: drug[batch[i]] += h[i]  (256 feats, vec4)
__global__ void k_pool4(const float* __restrict__ h, const int* __restrict__ batch,
                        float* drug, long long Nn) {
  long long t = (long long)blockIdx.x * blockDim.x + threadIdx.x;
  if (t >= Nn * 64) return;
  long long i = t >> 6; int c = ((int)(t & 63)) << 2;
  int b = batch[i];
  const float* hp = h + i * 256 + c;
  float*       dp = drug + (long long)b * 256 + c;
  atomicAdd(dp + 0, hp[0]); atomicAdd(dp + 1, hp[1]);
  atomicAdd(dp + 2, hp[2]); atomicAdd(dp + 3, hp[3]);
}

// A builder: A[i,k] = f16(h[i,k] + agg[i,k]) for k<F else 0  (row stride Kpad)
__global__ void k_build_a(const float* __restrict__ h, const float* __restrict__ agg,
                          _Float16* __restrict__ A, long long Nn, int F, int Kpad) {
  long long t = (long long)blockIdx.x * blockDim.x + threadIdx.x;
  if (t >= Nn * Kpad) return;
  long long i = t / Kpad; int k = (int)(t % Kpad);
  float v = 0.f;
  if (k < F) { v = h[i * F + k]; if (agg) v += agg[i * F + k]; }
  A[t] = (_Float16)v;
}

// ---------------------------------------------------------------- WMMA GEMM
// C[M x Nc] = A[M x K](f16, lda=K) * Bt^T   where Bt is [Nc x K] f16 (pre-transposed).
// Block: 256 threads / 8 waves -> 128x64 tile.
// Waves: 4 row-groups x 2 col-groups; wave tile 32x32 = 4 WMMA accumulators.
// mode 0: relu(acc+bias)                    -> f16 outH
// mode 1: relu((acc+bias)*g*BN+b) (+resid)  -> f32 outF
// mode 2: acc+bias                          -> f32 outF
__global__ __launch_bounds__(256) void k_gemm(
    const _Float16* __restrict__ A, int lda,
    const _Float16* __restrict__ Bt,
    float* outF, _Float16* __restrict__ outH,
    const float* __restrict__ bias,
    const float* __restrict__ gamma, const float* __restrict__ beta,
    const float* resid,
    int M, int Nc, int K, int mode) {
  __shared__ _Float16 As[128 * 32];  // 8 KB A stage
  const int tid  = threadIdx.x;
  const int lane = tid & 31, w = tid >> 5;
  const int wr = w & 3, wc = w >> 2;
  const int rowBase = blockIdx.x * 128;
  const int colBase = blockIdx.y * 64 + wc * 32;
  const int mrow = lane & 15;
  const int hlf  = lane >> 4;

  v8f acc00 = {}, acc01 = {}, acc10 = {}, acc11 = {};

  // B fragment pointers: lane -> col n, 16 contiguous K values per step half
  const _Float16* bp0 = Bt + (size_t)(colBase + mrow) * lda + hlf * 16;
  const _Float16* bp1 = bp0 + (size_t)16 * lda;

  // cooperative A staging: 512 8-half chunks; thread handles chunks tid and tid+256
  const int c0r = tid >> 2,          c0k = (tid & 3) * 8;
  const int c1r = (tid + 256) >> 2,  c1k = c0k;
  int g0 = rowBase + c0r; if (g0 >= M) g0 = M - 1;
  int g1 = rowBase + c1r; if (g1 >= M) g1 = M - 1;
  const _Float16* gap0 = A + (size_t)g0 * lda + c0k;
  const _Float16* gap1 = A + (size_t)g1 * lda + c1k;
  _Float16* lp0 = &As[c0r * 32 + c0k];
  _Float16* lp1 = &As[c1r * 32 + c1k];

  // A fragment LDS pointers for the two 16-row sub-tiles of this wave
  const _Float16* af0 = &As[(wr * 32 + mrow) * 32 + hlf * 16];
  const _Float16* af1 = af0 + 16 * 32;

  for (int k0 = 0; k0 < K; k0 += 32) {
    __syncthreads();
#if HAVE_ASYNC_LDS
    __builtin_amdgcn_global_load_async_to_lds_b128(
        (v4i*)(void*)(gap0 + k0), (v4i*)(void*)lp0, 0, 0);
    __builtin_amdgcn_global_load_async_to_lds_b128(
        (v4i*)(void*)(gap1 + k0), (v4i*)(void*)lp1, 0, 0);
    asm volatile("s_wait_asynccnt 0" ::: "memory");
#else
    *(v8h*)lp0 = *(const v8h*)(gap0 + k0);
    *(v8h*)lp1 = *(const v8h*)(gap1 + k0);
#endif
    if (k0 + 32 < K) {
      __builtin_prefetch(gap0 + k0 + 32, 0, 1);
      __builtin_prefetch(gap1 + k0 + 32, 0, 1);
    }
    __syncthreads();

    // A fragments: contiguous k = 16*laneHalf + s, same order as B
    v8h a0lo = *(const v8h*)(af0);
    v8h a0hi = *(const v8h*)(af0 + 8);
    v8h a1lo = *(const v8h*)(af1);
    v8h a1hi = *(const v8h*)(af1 + 8);
    v16h a0, a1;
#pragma unroll
    for (int i = 0; i < 8; ++i) {
      a0[i] = a0lo[i]; a0[8 + i] = a0hi[i];
      a1[i] = a1lo[i]; a1[8 + i] = a1hi[i];
    }
    v16h b0 = *(const v16h*)(bp0 + k0);
    v16h b1 = *(const v16h*)(bp1 + k0);
    acc00 = __builtin_amdgcn_wmma_f32_16x16x32_f16(false, a0, false, b0, (short)0, acc00, false, false);
    acc01 = __builtin_amdgcn_wmma_f32_16x16x32_f16(false, a0, false, b1, (short)0, acc01, false, false);
    acc10 = __builtin_amdgcn_wmma_f32_16x16x32_f16(false, a1, false, b0, (short)0, acc10, false, false);
    acc11 = __builtin_amdgcn_wmma_f32_16x16x32_f16(false, a1, false, b1, (short)0, acc11, false, false);
  }

  // epilogue: VGPR v -> row (v + 8*laneHalf); col = lane%16 (+16 for *1 accs)
  const int col0 = colBase + mrow;
#pragma unroll
  for (int r = 0; r < 2; ++r) {
    const int rtile = wr * 32 + r * 16 + hlf * 8;
#pragma unroll
    for (int v = 0; v < 8; ++v) {
      int row = rowBase + rtile + v;
      if (row >= M) continue;
#pragma unroll
      for (int j = 0; j < 2; ++j) {
        int col = col0 + j * 16;
        float y = (r == 0 ? (j == 0 ? acc00[v] : acc01[v])
                          : (j == 0 ? acc10[v] : acc11[v])) + bias[col];
        size_t idx = (size_t)row * Nc + col;
        if (mode == 0) {
          y = fmaxf(y, 0.f);
          outH[idx] = (_Float16)y;
        } else if (mode == 1) {
          y = y * (gamma[col] * BN_FOLD) + beta[col];
          y = fmaxf(y, 0.f);
          if (resid) y += resid[idx];
          outF[idx] = y;
        } else {
          outF[idx] = y;
        }
      }
    }
  }
}

// LayerNorm(256) + ReLU -> cat[:, 256:512] f16. One wave per row (wave32).
__global__ void k_ln_relu(const float* __restrict__ p, const float* __restrict__ g,
                          const float* __restrict__ b, _Float16* __restrict__ cat, int B) {
  int row  = blockIdx.x * 8 + (threadIdx.x >> 5);
  int lane = threadIdx.x & 31;
  if (row >= B) return;
  const float* pr = p + (size_t)row * 256;
  float vals[8]; float s = 0.f, s2 = 0.f;
#pragma unroll
  for (int i = 0; i < 8; ++i) { float v = pr[lane * 8 + i]; vals[i] = v; s += v; s2 += v * v; }
#pragma unroll
  for (int m = 16; m >= 1; m >>= 1) { s += __shfl_xor(s, m, 32); s2 += __shfl_xor(s2, m, 32); }
  float mu = s * (1.f / 256.f);
  float var = s2 * (1.f / 256.f) - mu * mu;
  float inv = rsqrtf(var + 1e-5f);
#pragma unroll
  for (int i = 0; i < 8; ++i) {
    int c = lane * 8 + i;
    float y = (vals[i] - mu) * inv * g[c] + b[c];
    cat[(size_t)row * 512 + 256 + c] = (_Float16)fmaxf(y, 0.f);
  }
}

__global__ void k_drug_to_cat(const float* __restrict__ drug, _Float16* __restrict__ cat,
                              long long B) {
  long long t = (long long)blockIdx.x * blockDim.x + threadIdx.x;
  if (t >= B * 256) return;
  long long b = t >> 8; int f = (int)(t & 255);
  cat[b * 512 + f] = (_Float16)drug[t];
}

// final: out[row] = dot(h2[row, 0:512], w) + b0. One wave per row.
__global__ void k_final(const _Float16* __restrict__ h2, const float* __restrict__ w,
                        const float* __restrict__ b, float* out, int B) {
  int row  = blockIdx.x * 8 + (threadIdx.x >> 5);
  int lane = threadIdx.x & 31;
  if (row >= B) return;
  float s = 0.f;
#pragma unroll
  for (int i = 0; i < 16; ++i) {
    int c = lane * 16 + i;
    s += (float)h2[(size_t)row * 512 + c] * w[c];
  }
#pragma unroll
  for (int m = 16; m >= 1; m >>= 1) s += __shfl_xor(s, m, 32);
  if (lane == 0) out[row] = s + b[0];
}

// ---------------------------------------------------------------- launcher
extern "C" void kernel_launch(void* const* d_in, const int* in_sizes, int n_in,
                              void* d_out, int out_size, void* d_ws, size_t ws_size,
                              hipStream_t stream) {
  const float* x     = (const float*)d_in[0];
  const int*   ei    = (const int*)d_in[1];
  const int*   batch = (const int*)d_in[2];
  const float* prot  = (const float*)d_in[3];
  const float* g0w1 = (const float*)d_in[4];  const float* g0b1 = (const float*)d_in[5];
  const float* g0w2 = (const float*)d_in[6];  const float* g0b2 = (const float*)d_in[7];
  const float* g0ga = (const float*)d_in[8];  const float* g0be = (const float*)d_in[9];
  const float* gw1  = (const float*)d_in[10]; const float* gb1  = (const float*)d_in[11];
  const float* gw2  = (const float*)d_in[12]; const float* gb2  = (const float*)d_in[13];
  const float* gga  = (const float*)d_in[14]; const float* gbe  = (const float*)d_in[15];
  const float* wp   = (const float*)d_in[16]; const float* bp   = (const float*)d_in[17];
  const float* lng  = (const float*)d_in[18]; const float* lnb  = (const float*)d_in[19];
  const float* wpr1 = (const float*)d_in[20]; const float* bpr1 = (const float*)d_in[21];
  const float* wpr2 = (const float*)d_in[22]; const float* bpr2 = (const float*)d_in[23];
  const float* wpr3 = (const float*)d_in[24]; const float* bpr3 = (const float*)d_in[25];

  const long long N = in_sizes[0] / 70;
  const long long E = in_sizes[1] / 2;
  const long long B = in_sizes[3] / 480;
  const int* src = ei;
  const int* dst = ei + E;

  size_t off = 0;
  auto take = [&](size_t bytes) -> char* {
    char* p = (char*)d_ws + off;
    off += (bytes + 255) & ~(size_t)255;
    return p;
  };
  float*    h    = (float*)take((size_t)N * 256 * 4);
  float*    agg  = (float*)take((size_t)N * 256 * 4);
  _Float16* aF   = (_Float16*)take((size_t)N * 256 * 2);
  _Float16* zF   = (_Float16*)take((size_t)N * 256 * 2);
  _Float16* w01t = (_Float16*)take((size_t)256 * 96 * 2);
  _Float16* w02t = (_Float16*)take((size_t)256 * 256 * 2);
  _Float16* w1t  = (_Float16*)take((size_t)3 * 256 * 256 * 2);
  _Float16* w2t  = (_Float16*)take((size_t)3 * 256 * 256 * 2);
  _Float16* wpt  = (_Float16*)take((size_t)256 * 480 * 2);
  _Float16* wp1t = (_Float16*)take((size_t)1024 * 512 * 2);
  _Float16* wp2t = (_Float16*)take((size_t)512 * 1024 * 2);
  float*    drug = (float*)take((size_t)B * 256 * 4);
  float*    pbuf = (float*)take((size_t)B * 256 * 4);
  _Float16* catb = (_Float16*)take((size_t)B * 512 * 2);
  _Float16* h1   = (_Float16*)take((size_t)B * 1024 * 2);
  _Float16* h2   = (_Float16*)take((size_t)B * 512 * 2);
  (void)ws_size; (void)n_in; (void)out_size;

  auto cdiv = [](long long a, long long b) { return (unsigned)((a + b - 1) / b); };
  dim3 blk(256);

  // weight repack (fp32 KxN -> f16 NxKpad)
  k_wt<<<cdiv(256LL * 96, 256), blk, 0, stream>>>(g0w1, w01t, 70, 256, 96);
  k_wt<<<cdiv(256LL * 256, 256), blk, 0, stream>>>(g0w2, w02t, 256, 256, 256);
  for (int i = 0; i < 3; ++i) {
    k_wt<<<cdiv(256LL * 256, 256), blk, 0, stream>>>(gw1 + (size_t)i * 65536, w1t + (size_t)i * 65536, 256, 256, 256);
    k_wt<<<cdiv(256LL * 256, 256), blk, 0, stream>>>(gw2 + (size_t)i * 65536, w2t + (size_t)i * 65536, 256, 256, 256);
  }
  k_wt<<<cdiv(256LL * 480, 256), blk, 0, stream>>>(wp, wpt, 480, 256, 480);
  k_wt<<<cdiv(1024LL * 512, 256), blk, 0, stream>>>(wpr1, wp1t, 512, 1024, 512);
  k_wt<<<cdiv(512LL * 1024, 256), blk, 0, stream>>>(wpr2, wp2t, 1024, 512, 1024);

  // ---- GIN layer 0 (70 -> 256) ----
  k_zero<<<cdiv(N * 70, 256), blk, 0, stream>>>(agg, N * 70);
  k_scatter1<<<cdiv(E * 70, 256), blk, 0, stream>>>(x, src, dst, agg, E, 70);
  k_build_a<<<cdiv(N * 96, 256), blk, 0, stream>>>(x, agg, aF, N, 70, 96);
  {
    dim3 g(cdiv(N, 128), 4);
    k_gemm<<<g, blk, 0, stream>>>(aF, 96, w01t, nullptr, zF, g0b1,
                                  nullptr, nullptr, nullptr, (int)N, 256, 96, 0);
    k_gemm<<<g, blk, 0, stream>>>(zF, 256, w02t, h, nullptr, g0b2,
                                  g0ga, g0be, nullptr, (int)N, 256, 256, 1);
  }

  // ---- GIN layers 1..3 with residual ----
  for (int i = 0; i < 3; ++i) {
    k_zero<<<cdiv(N * 256, 256), blk, 0, stream>>>(agg, N * 256);
    k_scatter4<<<cdiv(E * 64, 256), blk, 0, stream>>>(h, src, dst, agg, E, 256);
    k_build_a<<<cdiv(N * 256, 256), blk, 0, stream>>>(h, agg, aF, N, 256, 256);
    dim3 g(cdiv(N, 128), 4);
    k_gemm<<<g, blk, 0, stream>>>(aF, 256, w1t + (size_t)i * 65536, nullptr, zF,
                                  gb1 + i * 256, nullptr, nullptr, nullptr, (int)N, 256, 256, 0);
    k_gemm<<<g, blk, 0, stream>>>(zF, 256, w2t + (size_t)i * 65536, h, nullptr,
                                  gb2 + i * 256, gga + i * 256, gbe + i * 256, h, (int)N, 256, 256, 1);
  }

  // ---- global_add_pool ----
  k_zero<<<cdiv(B * 256, 256), blk, 0, stream>>>(drug, B * 256);
  k_pool4<<<cdiv(N * 64, 256), blk, 0, stream>>>(h, batch, drug, N);

  // ---- protein projector ----
  k_build_a<<<cdiv(B * 480, 256), blk, 0, stream>>>(prot, nullptr, aF, B, 480, 480);
  {
    dim3 g(cdiv(B, 128), 4);
    k_gemm<<<g, blk, 0, stream>>>(aF, 480, wpt, pbuf, nullptr, bp,
                                  nullptr, nullptr, nullptr, (int)B, 256, 480, 2);
  }
  k_ln_relu<<<cdiv(B, 8), blk, 0, stream>>>(pbuf, lng, lnb, catb, (int)B);
  k_drug_to_cat<<<cdiv(B * 256, 256), blk, 0, stream>>>(drug, catb, B);

  // ---- predictor MLP ----
  {
    dim3 g(cdiv(B, 128), 16);
    k_gemm<<<g, blk, 0, stream>>>(catb, 512, wp1t, nullptr, h1, bpr1,
                                  nullptr, nullptr, nullptr, (int)B, 1024, 512, 0);
  }
  {
    dim3 g(cdiv(B, 128), 8);
    k_gemm<<<g, blk, 0, stream>>>(h1, 1024, wp2t, nullptr, h2, bpr2,
                                  nullptr, nullptr, nullptr, (int)B, 512, 1024, 0);
  }
  k_final<<<cdiv(B, 8), blk, 0, stream>>>(h2, wpr3, bpr3, (float*)d_out, (int)B);
}
